// CapsNet_2619930050941
// MI455X (gfx1250) — matrix operand order repaired and checked by reading, not compiled, not run on hardware
//
#include <hip/hip_runtime.h>
#include <hip/hip_bf16.h>
#include <stdint.h>

typedef __attribute__((ext_vector_type(16))) _Float16 v16h;
typedef __attribute__((ext_vector_type(8)))  float    v8f;

union AF { v16h v; _Float16 h[16]; uint32_t u[8]; uint4 q[2]; };
union CF { v8f  v; float f[8]; };
union KI { int4 q[4]; int i[16]; };

// ---------------------------------------------------------------------------
// Workspace layout (bytes, all 256-aligned)
//   Wsw  f16 [3][16][32][16]      conv1 weights, frag-swizzled           49,152
//   Bsw  f16 [648][16][32][16]    conv2 weights, frag-swizzled      10,616,832
//   koff i32 [20736+96]           im2col offset tables (conv2+conv1)    83,456
//   x16  f16 [512][256][20][20]   conv1 output (relu), f16         104,857,600
//   vbuf f32 [18432][256]         conv2 raw output                  18,874,368
//   po   f16 [512][1152][8]       primary caps (squashed)            9,437,184
//   vt   f16 [512][1152][160]     digit votes (fits 192MB L2)      188,743,680
//   lg   f32 [512][1152][10]      routing logits                    23,592,960
//   h1   f32 [512][512]                                              1,048,576
//   h2   f32 [512][1024]                                              2,097,152
// ---------------------------------------------------------------------------
#define O_WC   0
#define O_BP   49152
#define O_KOFF 10665984
#define O_X16  10749440
#define O_VB   115607040
#define O_PO   134481408
#define O_VT   143918592
#define O_LG   332662272
#define O_H1   356255232
#define O_H2   357303808

// Prepack: weights into the exact per-lane WMMA B-fragment layout so a frag
// load is two contiguous b128s; also build the im2col K->offset tables.
// Swizzle: sw[((kt*16+nt)*32 + lane)*16 + e] = W[k=kt*32+(lane>>4)*8+(e&7)+(e>>3)*16][col=nt*16+(lane&15)]
__global__ void k_prepack(const float* __restrict__ conv_w,
                          const float* __restrict__ prim_w,
                          _Float16* __restrict__ Wsw, _Float16* __restrict__ Bsw,
                          int* __restrict__ koff) {
  int idx = blockIdx.x * 256 + threadIdx.x;
  if (idx < 20736) {                       // conv2 im2col offsets (elements)
    int c = idx / 81; int r = idx - c * 81; int ky = r / 9; int kx = r - ky * 9;
    koff[idx] = c * 400 + ky * 20 + kx;
  }
  if (idx < 96) {                          // conv1 im2col offsets; pad -> 0 (x0 weight)
    int ky = idx / 9, kx = idx - ky * 9;
    koff[20736 + idx] = (idx < 81) ? (ky * 28 + kx) : 0;
  }
  if (idx < 24576) {                       // conv1 weights, K padded 81->96 with zeros
    int e = idx & 15, L = (idx >> 4) & 31, nt = (idx >> 9) & 15, ks = idx >> 13;
    int k   = ks * 32 + (L >> 4) * 8 + (e & 7) + (e >> 3) * 16;
    int col = nt * 16 + (L & 15);
    float v = (k < 81) ? conv_w[col * 81 + k] : 0.0f;
    Wsw[idx] = (_Float16)v;
  }
  if (idx < 5308416) {                     // conv2 weights
    int e = idx & 15, L = (idx >> 4) & 31, ng = (idx >> 9) & 15, kt = idx >> 13;
    int k   = kt * 32 + (L >> 4) * 8 + (e & 7) + (e >> 3) * 16;
    int col = ng * 16 + (L & 15);
    Bsw[idx] = (_Float16)prim_w[col * 20736 + k];
  }
}

// conv1 as implicit-im2col WMMA GEMM: M=512*400, N=256, K=96 (81 padded).
// One block = 4 waves; each wave: one A frag feeds 4 N-tiles (4 WMMA / k-step).
__global__ void __launch_bounds__(128) k_conv1(const float* __restrict__ input,
                                               const float* __restrict__ conv_b,
                                               const _Float16* __restrict__ Wsw,
                                               const int* __restrict__ koff1,
                                               _Float16* __restrict__ x16) {
  const int tid  = threadIdx.x;
  const int w    = tid >> 5;
  const int lane = tid & 31;
  const int m0   = blockIdx.x * 16;
  const int lr = lane & 15;
  const int hh = lane >> 4;
  const int h8 = hh * 8;

  const int m  = m0 + lr;
  const int b  = m / 400;
  const int s  = m - b * 400;
  const int oy = s / 20;
  const int ox = s - oy * 20;
  const int ibase = b * 784 + oy * 28 + ox;

  CF acc[4];
  #pragma unroll
  for (int f = 0; f < 4; ++f)
    #pragma unroll
    for (int r = 0; r < 8; ++r) acc[f].f[r] = 0.0f;

  #pragma unroll
  for (int ks = 0; ks < 3; ++ks) {
    const int k0 = ks * 32;
    KI kk;                                  // two runs of 8 consecutive K values
    const int4* kp0 = (const int4*)(koff1 + k0 + h8);
    const int4* kp1 = (const int4*)(koff1 + k0 + 16 + h8);
    kk.q[0] = kp0[0]; kk.q[1] = kp0[1];
    kk.q[2] = kp1[0]; kk.q[3] = kp1[1];
    AF bb[4];                               // preload all 4 weight frags first
    #pragma unroll
    for (int f = 0; f < 4; ++f) {
      const uint4* bp = (const uint4*)(Wsw + (((ks * 16 + (w * 4 + f)) * 32 + lane) << 4));
      bb[f].q[0] = bp[0];
      bb[f].q[1] = bp[1];
    }
    AF a;
    #pragma unroll
    for (int e = 0; e < 16; ++e) a.h[e] = (_Float16)input[ibase + kk.i[e]];
    #pragma unroll
    for (int f = 0; f < 4; ++f)
      acc[f].v = __builtin_amdgcn_wmma_f32_16x16x32_f16(false, a.v, false, bb[f].v,
                                                        (short)0, acc[f].v, false, false);
  }
  int sbase[8];
  #pragma unroll
  for (int r = 0; r < 8; ++r) {
    int mm  = m0 + h8 + r;
    int bb2 = mm / 400;
    int ss  = mm - bb2 * 400;
    int oyy = ss / 20, oxx = ss - oyy * 20;
    sbase[r] = bb2 * 102400 + oyy * 20 + oxx;
  }
  #pragma unroll
  for (int f = 0; f < 4; ++f) {
    const int   co   = (w * 4 + f) * 16 + lr;
    const float bias = conv_b[co];
    #pragma unroll
    for (int r = 0; r < 8; ++r) {
      float val = acc[f].f[r] + bias;
      val = val > 0.0f ? val : 0.0f;
      x16[sbase[r] + co * 400] = (_Float16)val;
    }
  }
}

// conv2 (primary caps) WMMA GEMM: M=18432, N=256, K=20736.
// Pure register pipeline: no LDS, no barriers. Each wave computes 32x128
// (2 A-frags x 8 B-frags = 16 WMMA per 32-K step). B frags are preloaded in
// groups of 4 (8x b128 per clause, one wait, 8 chained WMMAs).
__global__ void __launch_bounds__(128) k_conv2(const _Float16* __restrict__ x16,
                                               const _Float16* __restrict__ Bsw,
                                               const int* __restrict__ koff,
                                               float* __restrict__ vbuf) {
  const int tid  = threadIdx.x;
  const int gw   = blockIdx.x * 4 + (tid >> 5);  // 0..1151
  const int lane = tid & 31;
  const int tm = gw >> 1;                  // 0..575
  const int tn = gw & 1;
  const int m0 = tm * 32;
  const int n0 = tn * 128;
  const int lr = lane & 15;
  const int hh = lane >> 4;
  const int h8 = hh * 8;
  const int ngbase = tn * 8;

  // per-lane A rows (one row per lane; halves of the wave split the K window)
  const int m0r = m0 + lr, m1r = m0 + 16 + lr;
  const int b0 = m0r / 36, s0 = m0r - b0 * 36, oy0 = s0 / 6, ox0 = s0 - oy0 * 6;
  const int b1 = m1r / 36, s1 = m1r - b1 * 36, oy1 = s1 / 6, ox1 = s1 - oy1 * 6;
  const _Float16* ab0 = x16 + b0 * 102400 + oy0 * 40 + ox0 * 2;
  const _Float16* ab1 = x16 + b1 * 102400 + oy1 * 40 + ox1 * 2;

  CF acc[2][8];
  #pragma unroll
  for (int ti = 0; ti < 2; ++ti)
    #pragma unroll
    for (int f = 0; f < 8; ++f)
      #pragma unroll
      for (int r = 0; r < 8; ++r) acc[ti][f].f[r] = 0.0f;

  for (int kt = 0; kt < 648; ++kt) {
    const int k0 = kt * 32;
    KI kk;                                  // two runs of 8 consecutive K values
    const int4* kp0 = (const int4*)(koff + k0 + h8);
    const int4* kp1 = (const int4*)(koff + k0 + 16 + h8);
    kk.q[0] = kp0[0]; kk.q[1] = kp0[1];
    kk.q[2] = kp1[0]; kk.q[3] = kp1[1];

    AF a0, a1;
    #pragma unroll
    for (int e = 0; e < 16; ++e) {
      a0.h[e] = ab0[kk.i[e]];
      a1.h[e] = ab1[kk.i[e]];
    }
    if (kt + 2 < 648) {                     // gfx1250 prefetch of B stream
      __builtin_prefetch(Bsw + ((((size_t)(kt + 2) * 16 + ngbase) * 32 + lane) << 4), 0, 1);
    }
    #pragma unroll
    for (int g = 0; g < 2; ++g) {           // two groups of 4 B-frags
      AF bf[4];
      #pragma unroll
      for (int f = 0; f < 4; ++f) {
        const uint4* bp = (const uint4*)
            (Bsw + ((((size_t)kt * 16 + ngbase + g * 4 + f) * 32 + lane) << 4));
        bf[f].q[0] = bp[0];
        bf[f].q[1] = bp[1];
      }
      #pragma unroll
      for (int f = 0; f < 4; ++f) {
        acc[0][g * 4 + f].v = __builtin_amdgcn_wmma_f32_16x16x32_f16(
            false, a0.v, false, bf[f].v, (short)0, acc[0][g * 4 + f].v, false, false);
        acc[1][g * 4 + f].v = __builtin_amdgcn_wmma_f32_16x16x32_f16(
            false, a1.v, false, bf[f].v, (short)0, acc[1][g * 4 + f].v, false, false);
      }
    }
  }

  #pragma unroll
  for (int ti = 0; ti < 2; ++ti)
    #pragma unroll
    for (int f = 0; f < 8; ++f)
      #pragma unroll
      for (int r = 0; r < 8; ++r) {
        int row = m0 + ti * 16 + h8 + r;
        int col = n0 + f * 16 + lr;
        vbuf[row * 256 + col] = acc[ti][f].f[r];
      }
}

// Primary routing (iter=1, i=1): route = 1/32 constant -> pre = v/32 + bias,
// squash over the 8-dim, store as prim_out f16 [b][i=cap*36+s][n].
__global__ void k_prim_squash(const float* __restrict__ vbuf,
                              const float* __restrict__ prim_bias,
                              _Float16* __restrict__ po) {
  int idx = blockIdx.x * 256 + threadIdx.x;
  if (idx >= 512 * 32 * 36) return;
  int b   = idx / (32 * 36);
  int rem = idx - b * (32 * 36);
  int cap = rem / 36;
  int s   = rem - cap * 36;
  float p[8]; float n2 = 0.0f;
  #pragma unroll
  for (int n = 0; n < 8; ++n) {
    float v = vbuf[(b * 36 + s) * 256 + cap * 8 + n] * (1.0f / 32.0f)
            + prim_bias[cap * 8 + n];
    p[n] = v; n2 += v * v;
  }
  float sc = n2 / ((1.0f + n2) * sqrtf(n2 + 1e-9f));
  int i = cap * 36 + s;
  #pragma unroll
  for (int n = 0; n < 8; ++n)
    po[(b * 1152 + i) * 8 + n] = (_Float16)(p[n] * sc);
}

// Digit votes: votes[b,i,m] = sum_n po[b,i,n] * digit_w[i,n,m]; K=8 (trivial FLOPs).
// Stored f16 so the whole tensor (189 MB) fits in the 192 MB L2 for routing.
__global__ void k_votes(const _Float16* __restrict__ po,
                        const float* __restrict__ dw,
                        _Float16* __restrict__ votes) {
  __shared__ float sp[8];
  int blk = blockIdx.x;            // b*1152 + i
  int t   = threadIdx.x;           // 0..159
  if (t < 8) sp[t] = (float)po[blk * 8 + t];
  __syncthreads();
  int i = blk % 1152;
  float acc = 0.0f;
  #pragma unroll
  for (int n = 0; n < 8; ++n) acc += sp[n] * dw[i * 1280 + n * 160 + t];
  votes[blk * 160 + t] = (_Float16)acc;
}

// Dynamic routing, 3 iterations fused; one block per batch element.
__global__ void __launch_bounds__(256) k_routing(const _Float16* __restrict__ votes,
                                                 const float* __restrict__ digit_bias,
                                                 float* __restrict__ logits,
                                                 float* __restrict__ out_digit) {
  __shared__ float s_route[1152 * 10];
  __shared__ float s_pre[160];
  __shared__ float s_act[160];
  const int b = blockIdx.x;
  const int t = threadIdx.x;
  const _Float16* vb = votes + (size_t)b * 1152 * 160;
  float* lg = logits + (size_t)b * 1152 * 10;

  for (int it = 0; it < 3; ++it) {
    if (t < 160) {                           // pre[c,n] = sum_i route * votes
      int c = t >> 4;
      float acc = 0.0f;
      if (it == 0) {                         // softmax(0) over 10 classes = 0.1
        for (int i = 0; i < 1152; ++i) acc += (float)vb[i * 160 + t];
        acc *= 0.1f;
      } else {
        for (int i = 0; i < 1152; ++i)
          acc += s_route[i * 10 + c] * (float)vb[i * 160 + t];
      }
      s_pre[t] = acc + digit_bias[t];
    }
    __syncthreads();
    if (t < 160) {                           // squash over n=16
      int c = t >> 4;
      float n2 = 0.0f;
      #pragma unroll
      for (int n = 0; n < 16; ++n) { float v = s_pre[c * 16 + n]; n2 += v * v; }
      float sc = n2 / ((1.0f + n2) * sqrtf(n2 + 1e-9f));
      float a = s_pre[t] * sc;
      s_act[t] = a;
      out_digit[b * 160 + t] = a;            // final iteration's value persists
    }
    __syncthreads();
    if (it < 2) {                            // logits += votes . act; next route
      for (int i = t; i < 1152; i += 256) {
        float l[10];
        #pragma unroll
        for (int c = 0; c < 10; ++c) {
          float old = (it == 0) ? 0.0f : lg[i * 10 + c];
          float dot = 0.0f;
          #pragma unroll
          for (int n = 0; n < 16; ++n)
            dot += (float)vb[i * 160 + c * 16 + n] * s_act[c * 16 + n];
          l[c] = old + dot;
          lg[i * 10 + c] = l[c];
        }
        float mx = l[0];
        #pragma unroll
        for (int c = 1; c < 10; ++c) mx = fmaxf(mx, l[c]);
        float e10[10]; float sum = 0.0f;
        #pragma unroll
        for (int c = 0; c < 10; ++c) { e10[c] = __expf(l[c] - mx); sum += e10[c]; }
        float inv = 1.0f / sum;
        #pragma unroll
        for (int c = 0; c < 10; ++c) s_route[i * 10 + c] = e10[c] * inv;
      }
    }
    __syncthreads();
  }
}

// Decoder. fc1 exploits one-hot masking: only 16 of 160 inputs are nonzero.
__global__ void k_fc1(const float* __restrict__ out_digit, const int* __restrict__ labels,
                      const float* __restrict__ w, const float* __restrict__ bias,
                      float* __restrict__ h1) {
  int idx = blockIdx.x * 256 + threadIdx.x;
  if (idx >= 512 * 512) return;
  int b = idx >> 9, o = idx & 511;
  int lab = labels[b];
  float acc = bias[o];
  #pragma unroll
  for (int n = 0; n < 16; ++n)
    acc += out_digit[b * 160 + lab * 16 + n] * w[(lab * 16 + n) * 512 + o];
  h1[idx] = acc > 0.0f ? acc : 0.0f;
}

__global__ void k_fc2(const float* __restrict__ h1, const float* __restrict__ w,
                      const float* __restrict__ bias, float* __restrict__ h2) {
  int idx = blockIdx.x * 256 + threadIdx.x;
  if (idx >= 512 * 1024) return;
  int b = idx >> 10, o = idx & 1023;
  float acc = bias[o];
  for (int k = 0; k < 512; ++k) acc += h1[b * 512 + k] * w[k * 1024 + o];
  h2[idx] = acc > 0.0f ? acc : 0.0f;
}

__global__ void k_fc3(const float* __restrict__ h2, const float* __restrict__ w,
                      const float* __restrict__ bias, float* __restrict__ recon) {
  int idx = blockIdx.x * 256 + threadIdx.x;
  if (idx >= 512 * 784) return;
  int b = idx / 784, p = idx - b * 784;
  float acc = bias[p];
  for (int k = 0; k < 1024; ++k) acc += h2[b * 1024 + k] * w[k * 784 + p];
  recon[idx] = 1.0f / (1.0f + __expf(-acc));
}

extern "C" void kernel_launch(void* const* d_in, const int* in_sizes, int n_in,
                              void* d_out, int out_size, void* d_ws, size_t ws_size,
                              hipStream_t stream) {
  const float* input      = (const float*)d_in[0];
  const int*   labels     = (const int*)  d_in[1];
  const float* conv_w     = (const float*)d_in[2];
  const float* conv_b     = (const float*)d_in[3];
  const float* prim_w     = (const float*)d_in[4];
  const float* prim_bias  = (const float*)d_in[5];
  const float* digit_w    = (const float*)d_in[6];
  const float* digit_bias = (const float*)d_in[7];
  const float* fc1_w      = (const float*)d_in[8];
  const float* fc1_b      = (const float*)d_in[9];
  const float* fc2_w      = (const float*)d_in[10];
  const float* fc2_b      = (const float*)d_in[11];
  const float* fc3_w      = (const float*)d_in[12];
  const float* fc3_b      = (const float*)d_in[13];

  float* out = (float*)d_out;
  char*  ws  = (char*)d_ws;
  _Float16* Wsw = (_Float16*)(ws + O_WC);
  _Float16* Bsw = (_Float16*)(ws + O_BP);
  int*      kof = (int*)     (ws + O_KOFF);
  _Float16* x16 = (_Float16*)(ws + O_X16);
  float*    vb  = (float*)   (ws + O_VB);
  _Float16* po  = (_Float16*)(ws + O_PO);
  _Float16* vt  = (_Float16*)(ws + O_VT);
  float*    lg  = (float*)   (ws + O_LG);
  float*    h1  = (float*)   (ws + O_H1);
  float*    h2  = (float*)   (ws + O_H2);
  float* recon  = out + 512 * 160;

  hipLaunchKernelGGL(k_prepack,     dim3(20736),  dim3(256), 0, stream, conv_w, prim_w, Wsw, Bsw, kof);
  hipLaunchKernelGGL(k_conv1,       dim3(12800),  dim3(128), 0, stream, input, conv_b, Wsw, kof + 20736, x16);
  hipLaunchKernelGGL(k_conv2,       dim3(288),    dim3(128), 0, stream, x16, Bsw, kof, vb);
  hipLaunchKernelGGL(k_prim_squash, dim3(2304),   dim3(256), 0, stream, vb, prim_bias, po);
  hipLaunchKernelGGL(k_votes,       dim3(589824), dim3(160), 0, stream, po, digit_w, vt);
  hipLaunchKernelGGL(k_routing,     dim3(512),    dim3(256), 0, stream, vt, digit_bias, lg, out);
  hipLaunchKernelGGL(k_fc1,         dim3(1024),   dim3(256), 0, stream, out, labels, fc1_w, fc1_b, h1);
  hipLaunchKernelGGL(k_fc2,         dim3(2048),   dim3(256), 0, stream, h1, fc2_w, fc2_b, h2);
  hipLaunchKernelGGL(k_fc3,         dim3(1568),   dim3(256), 0, stream, h2, fc3_w, fc3_b, recon);
}
